// GNNFeatureExtractor_44976897524573
// MI455X (gfx1250) — matrix operand order repaired and checked by reading, not compiled, not run on hardware
//
#include <hip/hip_runtime.h>
#include <hip/hip_bf16.h>
#include <math.h>

typedef __bf16 bf16_t;
typedef __attribute__((ext_vector_type(16))) __bf16 v16bf;
typedef __attribute__((ext_vector_type(8)))  float  v8f;

#define HDIM 64
#define RE 1.0f            /* 2 / LAMBDA_MAX, LAMBDA_MAX = 2.0 */

// ---------------------------------------------------------------------------
// A-fragment loader (bf16 in, per CDNA5 ISA 7.12.2 16-bit A 16x32 layout):
//   lane(0..15) = row M, VGPR pair p<4 -> K = 2p,2p+1 ; p>=4 -> K = 8+2p,...
//   lanes 16..31 add +8 within each 16-K group.
// C and kofs are compile-time -> bounds checks fold away; loads collapse into
// global_load_b128 pairs + v_cvt_pk_bf16_f32. Caller guarantees row < N.
// ---------------------------------------------------------------------------
template <int C>
static __device__ __forceinline__ v16bf
load_a(const float* __restrict__ X, int row, int kofs, int khalf) {
  v16bf a;
  const float* xp = X + (long long)row * C;
#pragma unroll
  for (int p = 0; p < 8; ++p) {
    const int kbase = (p < 4) ? (2 * p) : (8 + 2 * p);
    if (kofs + kbase >= C) {           // compile-time (whole 16-K group OOB)
      a[2 * p]     = (bf16_t)0.0f;
      a[2 * p + 1] = (bf16_t)0.0f;
    } else {                            // full pair in range for C in {16,64}
      float2 x = *(const float2*)(xp + kofs + kbase + 8 * khalf);
      a[2 * p]     = (bf16_t)x.x;
      a[2 * p + 1] = (bf16_t)x.y;
    }
  }
  return a;
}

// ---------------------------------------------------------------------------
// Weight pre-pack: W[NW][C][64] (fp32, row-major) -> bf16 B-fragments laid out
// exactly as the WMMA B operand wants them:
//   P[((w*4 + ct)*KS + ks)*32 + lane][0..15], B-layout K = ks*32 + (p<4 ? 2p : 8+2p) + 8*khalf,
//   col = ct*16 + (lane&15).  One aligned 32-byte load per fragment in the GEMM.
// ---------------------------------------------------------------------------
__global__ void k_pack_w(const float* __restrict__ W, bf16_t* __restrict__ P,
                         int C, int total) {
  int idx = blockIdx.x * blockDim.x + threadIdx.x;
  if (idx >= total) return;
  const int ksteps = (C + 31) >> 5;
  const int p    = idx & 7;
  const int lane = (idx >> 3) & 31;
  int rest = idx >> 8;                 // ((w*4 + ct)*ksteps + ks)
  const int ks = rest % ksteps;  rest /= ksteps;
  const int ct = rest & 3;
  const int w  = rest >> 2;
  const int khalf = lane >> 4;
  const int col = ct * 16 + (lane & 15);
  const int k = ks * 32 + ((p < 4) ? (2 * p) : (8 + 2 * p)) + 8 * khalf;
  const float* Ww = W + (long long)w * C * HDIM;
  float f0 = (k     < C) ? Ww[(long long)k       * HDIM + col] : 0.0f;
  float f1 = (k + 1 < C) ? Ww[(long long)(k + 1) * HDIM + col] : 0.0f;
  long long o = (long long)(idx >> 3) * 16 + 2 * p;
  P[o]     = (bf16_t)f0;
  P[o + 1] = (bf16_t)f1;
}

// ---------------------------------------------------------------------------
// Fused ChebConv GEMM: out = relu(X0@W[0] + X1@W[1] + X2@W[2] + b)
// Block = 256 threads = 8 waves; each wave owns a 16x16 output tile
// (2 row-tiles x 4 col-tiles per block -> 32 rows x 64 cols).
// Epilogue: branch-free, one base pointer + constant offsets (N % 16 == 0).
// ---------------------------------------------------------------------------
template <int C>
__global__ void k_cheb_gemm(const float* __restrict__ X0, const float* __restrict__ X1,
                            const float* __restrict__ X2, const bf16_t* __restrict__ P,
                            const float* __restrict__ bias, float* __restrict__ out,
                            int N) {
  constexpr int KS = (C + 31) / 32;
  const int lane   = threadIdx.x & 31;
  const int wave   = threadIdx.x >> 5;
  const int khalf  = lane >> 4;
  const int lane16 = lane & 15;
  const int rowbase = (blockIdx.x * 2 + (wave >> 2)) * 16;
  if (rowbase >= N) return;
  const int ct   = wave & 3;
  const int col  = ct * 16 + lane16;
  const int rowA = rowbase + lane16;
  const v16bf* Pv = (const v16bf*)P;

  v8f acc = {};
  const float* Xs[3] = {X0, X1, X2};
#pragma unroll
  for (int w = 0; w < 3; ++w) {
#pragma unroll
    for (int ks = 0; ks < KS; ++ks) {
      v16bf a = load_a<C>(Xs[w], rowA, ks * 32, khalf);
      v16bf b = Pv[((w * 4 + ct) * KS + ks) * 32 + lane];
      acc = __builtin_amdgcn_wmma_f32_16x16x32_bf16(false, a, false, b,
                                                    (short)0, acc, false, false);
    }
  }
  const float bv = bias[col];
  // D layout: element v -> row = rowbase + v + 8*khalf, col = ct*16 + (lane&15)
  float* op = out + (long long)(rowbase + 8 * khalf) * HDIM + col;
#pragma unroll
  for (int v = 0; v < 8; ++v)
    op[v * HDIM] = fmaxf(acc[v] + bv, 0.0f);
}

// ---------------------------------------------------------------------------
// Fused EdgeConv precompute: u = X@Wt ; phi = X@Wp + bp   (C = 64, shared A)
// ---------------------------------------------------------------------------
__global__ void k_edge_gemm(const float* __restrict__ X, const bf16_t* __restrict__ Pt,
                            const bf16_t* __restrict__ Pp, const float* __restrict__ bp,
                            float* __restrict__ u, float* __restrict__ phi, int N) {
  const int lane   = threadIdx.x & 31;
  const int wave   = threadIdx.x >> 5;
  const int khalf  = lane >> 4;
  const int lane16 = lane & 15;
  const int rowbase = (blockIdx.x * 2 + (wave >> 2)) * 16;
  if (rowbase >= N) return;
  const int ct   = wave & 3;
  const int col  = ct * 16 + lane16;
  const int rowA = rowbase + lane16;
  const v16bf* Ptv = (const v16bf*)Pt;
  const v16bf* Ppv = (const v16bf*)Pp;

  v8f accU = {}, accP = {};
#pragma unroll
  for (int ks = 0; ks < 2; ++ks) {
    v16bf a  = load_a<HDIM>(X, rowA, ks * 32, khalf);
    v16bf bt = Ptv[(ct * 2 + ks) * 32 + lane];
    v16bf bb = Ppv[(ct * 2 + ks) * 32 + lane];
    accU = __builtin_amdgcn_wmma_f32_16x16x32_bf16(false, a, false, bt,
                                                   (short)0, accU, false, false);
    accP = __builtin_amdgcn_wmma_f32_16x16x32_bf16(false, a, false, bb,
                                                   (short)0, accP, false, false);
  }
  const float bv = bp[col];
  const long long base = (long long)(rowbase + 8 * khalf) * HDIM + col;
  float* up = u + base;
  float* pp = phi + base;
#pragma unroll
  for (int v = 0; v < 8; ++v) {
    up[v * HDIM] = accU[v];
    pp[v * HDIM] = accP[v] + bv;
  }
}

// ---------------------------------------------------------------------------
// Elementwise / edge kernels
// ---------------------------------------------------------------------------
__global__ void k_fill(float* __restrict__ p, long long n, float v) {
  long long i = (long long)blockIdx.x * blockDim.x + threadIdx.x;
  if (i < n) p[i] = v;
}

__global__ void k_deg(const int* __restrict__ dst, float* __restrict__ deg, int E) {
  int e = blockIdx.x * blockDim.x + threadIdx.x;
  if (e < E) atomicAdd(&deg[dst[e]], 1.0f);
}

__global__ void k_dinv(float* __restrict__ dinv, int N) {
  int i = blockIdx.x * blockDim.x + threadIdx.x;
  if (i < N) {
    float d = dinv[i];
    d = (d < 1.0f) ? 1.0f : d;
    dinv[i] = 1.0f / sqrtf(d);
  }
}

// m[dst] += X[src] * dinv[src]; one thread = one edge x 4 channels (float4)
__global__ void k_cheb_scatter(const int* __restrict__ src, const int* __restrict__ dst,
                               const float* __restrict__ X, const float* __restrict__ dinv,
                               float* __restrict__ m, long long total, int C, int shift) {
  long long idx = (long long)blockIdx.x * blockDim.x + threadIdx.x;
  if (idx >= total) return;
  int e  = (int)(idx >> shift);
  int c4 = ((int)idx & ((1 << shift) - 1)) << 2;
  int s = src[e], d = dst[e];
  float w = dinv[s];
  float4 x = *(const float4*)(X + (long long)s * C + c4);
  float* mp = m + (long long)d * C + c4;
  atomicAdd(mp + 0, x.x * w);
  atomicAdd(mp + 1, x.y * w);
  atomicAdd(mp + 2, x.z * w);
  atomicAdd(mp + 3, x.w * w);
}

__global__ void k_cheb_x1(const float* __restrict__ m, const float* __restrict__ X0,
                          const float* __restrict__ dinv, float* __restrict__ X1,
                          long long total, int nodeShift) {
  long long i = (long long)blockIdx.x * blockDim.x + threadIdx.x;
  if (i >= total) return;
  int node = (int)(i >> nodeShift);
  X1[i] = -RE * dinv[node] * m[i] + (RE - 1.0f) * X0[i];
}

__global__ void k_cheb_x2(const float* __restrict__ m, const float* __restrict__ X0,
                          const float* __restrict__ X1, const float* __restrict__ dinv,
                          float* __restrict__ X2, long long total, int nodeShift) {
  long long i = (long long)blockIdx.x * blockDim.x + threadIdx.x;
  if (i >= total) return;
  int node = (int)(i >> nodeShift);
  X2[i] = -2.0f * RE * dinv[node] * m[i] + 2.0f * (RE - 1.0f) * X1[i] - X0[i];
}

// float atomic max via monotone integer-bits mapping (init must be -INF)
static __device__ __forceinline__ void atomicMaxF(float* addr, float val) {
  if (val >= 0.0f) atomicMax((int*)addr, __float_as_int(val));
  else             atomicMin((unsigned int*)addr, (unsigned int)__float_as_int(val));
}

// agg[dst] = max(agg[dst], u[dst]-u[src]+phi[dst]+bt); one thread = edge x 4 ch
__global__ void k_edge_max(const int* __restrict__ src, const int* __restrict__ dst,
                           const float* __restrict__ u, const float* __restrict__ phi,
                           const float* __restrict__ bt, float* __restrict__ agg,
                           long long total) {
  long long idx = (long long)blockIdx.x * blockDim.x + threadIdx.x;
  if (idx >= total) return;
  int e  = (int)(idx >> 4);
  int c4 = ((int)idx & 15) << 2;
  int s = src[e], d = dst[e];
  float4 ud = *(const float4*)(u   + (long long)d * HDIM + c4);
  float4 us = *(const float4*)(u   + (long long)s * HDIM + c4);
  float4 ph = *(const float4*)(phi + (long long)d * HDIM + c4);
  float4 bb = *(const float4*)(bt + c4);
  float* ap = agg + (long long)d * HDIM + c4;
  atomicMaxF(ap + 0, ud.x - us.x + ph.x + bb.x);
  atomicMaxF(ap + 1, ud.y - us.y + ph.y + bb.y);
  atomicMaxF(ap + 2, ud.z - us.z + ph.z + bb.z);
  atomicMaxF(ap + 3, ud.w - us.w + ph.w + bb.w);
}

// relu finalize: handles "no in-edges" (-INF -> 0) and isfinite masking together
__global__ void k_relu(const float* __restrict__ a, float* __restrict__ o, long long n) {
  long long i = (long long)blockIdx.x * blockDim.x + threadIdx.x;
  if (i < n) o[i] = fmaxf(a[i], 0.0f);
}

__global__ void k_pool_cnt(const int* __restrict__ gid, float* __restrict__ gcnt, int N) {
  int i = blockIdx.x * blockDim.x + threadIdx.x;
  if (i < N) atomicAdd(&gcnt[gid[i]], 1.0f);
}

__global__ void k_pool_sum(const float* __restrict__ h, const int* __restrict__ gid,
                           float* __restrict__ gsum, long long total) {
  long long i = (long long)blockIdx.x * blockDim.x + threadIdx.x;
  if (i >= total) return;
  int node = (int)(i >> 6);
  int c = (int)i & 63;
  atomicAdd(&gsum[(long long)gid[node] * HDIM + c], h[i]);
}

__global__ void k_pool_final(const float* __restrict__ gsum, const float* __restrict__ gcnt,
                             float* __restrict__ out, int total) {
  int i = blockIdx.x * blockDim.x + threadIdx.x;
  if (i < total) {
    float c = gcnt[i >> 6];
    out[i] = gsum[i] / ((c < 1.0f) ? 1.0f : c);
  }
}

// ---------------------------------------------------------------------------
// Host orchestration
// ---------------------------------------------------------------------------
static inline unsigned cdivll(long long a, int b) { return (unsigned)((a + b - 1) / b); }

static void run_cheb(const float* Xin, int C, const bf16_t* P, const float* b,
                     const int* src, const int* dst, const float* dinv,
                     float* X1, float* X2, float* m, float* hout,
                     int N, int E, hipStream_t stream) {
  const long long nc = (long long)N * C;
  const int shift = (C == 16) ? 2 : 4;        // channels/4 per edge-thread
  const int nodeShift = (C == 16) ? 4 : 6;
  const long long etot = (long long)E << shift;
  const unsigned gemmGrid = (unsigned)((N + 31) / 32);

  k_fill<<<cdivll(nc, 256), 256, 0, stream>>>(m, nc, 0.0f);
  k_cheb_scatter<<<cdivll(etot, 256), 256, 0, stream>>>(src, dst, Xin, dinv, m, etot, C, shift);
  k_cheb_x1<<<cdivll(nc, 256), 256, 0, stream>>>(m, Xin, dinv, X1, nc, nodeShift);
  k_fill<<<cdivll(nc, 256), 256, 0, stream>>>(m, nc, 0.0f);
  k_cheb_scatter<<<cdivll(etot, 256), 256, 0, stream>>>(src, dst, X1, dinv, m, etot, C, shift);
  k_cheb_x2<<<cdivll(nc, 256), 256, 0, stream>>>(m, Xin, X1, dinv, X2, nc, nodeShift);
  if (C == 16)
    k_cheb_gemm<16><<<gemmGrid, 256, 0, stream>>>(Xin, X1, X2, P, b, hout, N);
  else
    k_cheb_gemm<64><<<gemmGrid, 256, 0, stream>>>(Xin, X1, X2, P, b, hout, N);
}

static void run_edgeconv(const float* h, const bf16_t* Pt, const float* bt,
                         const bf16_t* Pp, const float* bp,
                         const int* src, const int* dst,
                         float* u, float* phi, float* agg, float* hout,
                         int N, int E, hipStream_t stream) {
  const long long nh = (long long)N * HDIM;
  const long long etot = (long long)E << 4;
  const unsigned gemmGrid = (unsigned)((N + 31) / 32);
  k_edge_gemm<<<gemmGrid, 256, 0, stream>>>(h, Pt, Pp, bp, u, phi, N);
  k_fill<<<cdivll(nh, 256), 256, 0, stream>>>(agg, nh, -INFINITY);
  k_edge_max<<<cdivll(etot, 256), 256, 0, stream>>>(src, dst, u, phi, bt, agg, etot);
  k_relu<<<cdivll(nh, 256), 256, 0, stream>>>(agg, hout, nh);
}

extern "C" void kernel_launch(void* const* d_in, const int* in_sizes, int n_in,
                              void* d_out, int out_size, void* d_ws, size_t ws_size,
                              hipStream_t stream) {
  (void)n_in; (void)ws_size;
  const float* feat = (const float*)d_in[0];
  const int*   src  = (const int*)d_in[1];
  const int*   dst  = (const int*)d_in[2];
  const int*   gid  = (const int*)d_in[3];
  const float* W1 = (const float*)d_in[4];   const float* b1  = (const float*)d_in[5];
  const float* W2 = (const float*)d_in[6];   const float* b2  = (const float*)d_in[7];
  const float* W3 = (const float*)d_in[8];   const float* b3  = (const float*)d_in[9];
  const float* Wt1 = (const float*)d_in[10]; const float* bt1 = (const float*)d_in[11];
  const float* Wp1 = (const float*)d_in[12]; const float* bp1 = (const float*)d_in[13];
  const float* Wt2 = (const float*)d_in[14]; const float* bt2 = (const float*)d_in[15];
  const float* Wp2 = (const float*)d_in[16]; const float* bp2 = (const float*)d_in[17];

  const int IN = in_sizes[4] / (3 * HDIM);   // 16
  const int N  = in_sizes[0] / IN;           // 100000
  const int E  = in_sizes[1];                // 1600000
  const int G  = out_size / HDIM;            // 100

  // Packed-weight sizes (bf16 elems): NW * 4(coltiles) * KS * 32(lanes) * 16
  const int KS16 = 1, KS64 = 2;
  const int szP1 = 3 * 4 * KS16 * 32 * 16;   // 6144
  const int szP2 = 3 * 4 * KS64 * 32 * 16;   // 12288
  const int szPe = 1 * 4 * KS64 * 32 * 16;   // 4096

  // Workspace layout: [packed bf16 weights] [dinv N] [gcnt G] [gsum G*64] [B0..B4]
  bf16_t* Pbase = (bf16_t*)d_ws;
  bf16_t* P1  = Pbase;
  bf16_t* P2  = P1 + szP1;
  bf16_t* P3  = P2 + szP2;
  bf16_t* Pt1 = P3 + szP2;
  bf16_t* Pp1 = Pt1 + szPe;
  bf16_t* Pt2 = Pp1 + szPe;
  bf16_t* Pp2 = Pt2 + szPe;
  long long packedFloats = (szP1 + 2 * szP2 + 4 * szPe + 1) / 2;

  float* dinv = (float*)d_ws + packedFloats;
  float* gcnt = dinv + N;
  float* gsum = gcnt + G;
  float* B0 = gsum + (long long)G * HDIM;
  float* B1 = B0 + (long long)N * HDIM;
  float* B2 = B1 + (long long)N * HDIM;
  float* B3 = B2 + (long long)N * HDIM;
  float* B4 = B3 + (long long)N * HDIM;

  // Pack all weights into WMMA B-fragment layout (tiny, L2-resident)
  k_pack_w<<<cdivll(szP1 / 2, 256), 256, 0, stream>>>(W1,  P1,  IN,   szP1 / 2);
  k_pack_w<<<cdivll(szP2 / 2, 256), 256, 0, stream>>>(W2,  P2,  HDIM, szP2 / 2);
  k_pack_w<<<cdivll(szP2 / 2, 256), 256, 0, stream>>>(W3,  P3,  HDIM, szP2 / 2);
  k_pack_w<<<cdivll(szPe / 2, 256), 256, 0, stream>>>(Wt1, Pt1, HDIM, szPe / 2);
  k_pack_w<<<cdivll(szPe / 2, 256), 256, 0, stream>>>(Wp1, Pp1, HDIM, szPe / 2);
  k_pack_w<<<cdivll(szPe / 2, 256), 256, 0, stream>>>(Wt2, Pt2, HDIM, szPe / 2);
  k_pack_w<<<cdivll(szPe / 2, 256), 256, 0, stream>>>(Wp2, Pp2, HDIM, szPe / 2);

  // Degrees -> D^{-1/2}
  k_fill<<<cdivll(N, 256), 256, 0, stream>>>(dinv, N, 0.0f);
  k_deg<<<cdivll(E, 256), 256, 0, stream>>>(dst, dinv, E);
  k_dinv<<<cdivll(N, 256), 256, 0, stream>>>(dinv, N);

  // Layer 1: ChebConv(16 -> 64) + relu               (out: B0)
  run_cheb(feat, IN, P1, b1, src, dst, dinv, B1, B2, B3, B0, N, E, stream);
  // Layer 2: EdgeConv + relu                          (out: B4)
  run_edgeconv(B0, Pt1, bt1, Pp1, bp1, src, dst, B1, B2, B3, B4, N, E, stream);
  // Layer 3: ChebConv(64 -> 64) + relu                (out: B0)
  run_cheb(B4, HDIM, P2, b2, src, dst, dinv, B1, B2, B3, B0, N, E, stream);
  // Layer 4: EdgeConv + relu                          (out: B4)
  run_edgeconv(B0, Pt2, bt2, Pp2, bp2, src, dst, B1, B2, B3, B4, N, E, stream);
  // Layer 5: ChebConv(64 -> 64) + relu                (out: B0)
  run_cheb(B4, HDIM, P3, b3, src, dst, dinv, B1, B2, B3, B0, N, E, stream);

  // Per-graph mean pooling
  const long long pg = (long long)G * (HDIM + 1);
  k_fill<<<cdivll(pg, 256), 256, 0, stream>>>(gcnt, pg, 0.0f);
  k_pool_cnt<<<cdivll(N, 256), 256, 0, stream>>>(gid, gcnt, N);
  k_pool_sum<<<cdivll((long long)N * HDIM, 256), 256, 0, stream>>>(B0, gid, gsum, (long long)N * HDIM);
  k_pool_final<<<cdivll(G * HDIM, 256), 256, 0, stream>>>(gsum, gcnt, (float*)d_out, G * HDIM);
}
